// VisualMamba_86217173500356
// MI455X (gfx1250) — matrix operand order repaired and checked by reading, not compiled, not run on hardware
//
#include <hip/hip_runtime.h>
#include <math.h>

// ---------------- model dims ----------------
#define D_     384
#define E_     768
#define N_     16
#define EN_    12288
#define L_     2
#define B_     8
#define M_     197
#define RREAL  1576      // B_*M_
#define RPAD   1584      // 99*16
#define PTOK   1568      // B_*196  (98*16 exactly)
#define KPATCH 768       // 3*16*16

// ---------------- WMMA types ----------------
typedef __attribute__((ext_vector_type(16))) __bf16       v16bf;
typedef __attribute__((ext_vector_type(8)))  float        v8f;
typedef __attribute__((ext_vector_type(4)))  unsigned int u32x4;

union Frag { u32x4 q[2]; v16bf v; };

__device__ __forceinline__ unsigned short f2bf(float f) {
    unsigned int u = __float_as_uint(f);
    u += 0x7fffu + ((u >> 16) & 1u);          // round-to-nearest-even
    return (unsigned short)(u >> 16);
}
__device__ __forceinline__ float bf2f(unsigned short s) {
    return __uint_as_float(((unsigned int)s) << 16);
}

// Fragment loaders per CDNA5 16-bit A(16x32)/B(32x16) VGPR layouts.
__device__ __forceinline__ void load_stage(Frag& a, Frag b[4],
                                           const unsigned short* __restrict__ arow,
                                           const unsigned short* const wrow[4], int k)
{
    a.q[0] = *(const u32x4*)(arow + k);
    a.q[1] = *(const u32x4*)(arow + k + 16);
#pragma unroll
    for (int j = 0; j < 4; ++j) {
        b[j].q[0] = *(const u32x4*)(wrow[j] + k);
        b[j].q[1] = *(const u32x4*)(wrow[j] + k + 16);
    }
}

__device__ __forceinline__ void mma_stage(v8f acc[4], const Frag& a, const Frag b[4])
{
#pragma unroll
    for (int j = 0; j < 4; ++j)
        acc[j] = __builtin_amdgcn_wmma_f32_16x16x32_bf16(
            false, a.v, false, b[j].v, (short)0, acc[j], false, false);
}

// ---------------------------------------------------------------------------
// Generic bf16 WMMA GEMM:  out[m,n] = sum_k A[m,k]*W[n,k] + bias[n]
// A: (Mt*16, K) bf16 row-major;  W: (Nt*64, K) bf16 row-major.
// One wave computes a 16x64 tile (4 accumulators, A-frag reused 4x).
// Software-pipelined 2-stage ping-pong: next stage's 10 b128 loads are in
// flight while the current stage's 4 WMMAs execute (partial loadcnt waits).
// Requires K % 64 == 0 and K >= 128 (true for all GEMMs here).
// ---------------------------------------------------------------------------
__global__ __launch_bounds__(256)
void k_gemm_wmma(const unsigned short* __restrict__ A,
                 const unsigned short* __restrict__ W,
                 const float* __restrict__ bias,
                 float* __restrict__ outF,
                 unsigned short* __restrict__ outB,
                 int Mt, int Nt, int K)
{
    const int wave = (int)((blockIdx.x * blockDim.x + threadIdx.x) >> 5);
    if (wave >= Mt * Nt) return;
    const int lane = (int)(threadIdx.x & 31);
    const int mt = wave / Nt, nt = wave % Nt;
    const int hf  = lane >> 4;     // half-wave select
    const int l15 = lane & 15;
    const int Ncols = Nt * 64;

    // A: lane holds row (mt*16+l15); chunks at k+hf*8 and k+hf*8+16
    const unsigned short* arow = A + (size_t)(mt * 16 + l15) * K + hf * 8;
    // B (from W rows): lane holds col l15; chunks at k+hf*8 and k+hf*8+16
    const unsigned short* wrow[4];
#pragma unroll
    for (int j = 0; j < 4; ++j)
        wrow[j] = W + (size_t)(nt * 64 + j * 16 + l15) * K + hf * 8;

    v8f acc[4] = {};
    Frag a0, b0[4], a1, b1[4];

    load_stage(a0, b0, arow, wrow, 0);
    int k = 0;
    for (; k + 128 <= K; k += 64) {
        load_stage(a1, b1, arow, wrow, k + 32);
        __builtin_prefetch(arow + k + 256, 0, 3);
        mma_stage(acc, a0, b0);
        load_stage(a0, b0, arow, wrow, k + 64);
        mma_stage(acc, a1, b1);
    }
    // tail: final 64 of K (K % 64 == 0)
    load_stage(a1, b1, arow, wrow, k + 32);
    mma_stage(acc, a0, b0);
    mma_stage(acc, a1, b1);

    // C/D layout: VGPR g -> row = mt*16 + g + hf*8, col = nt*64 + j*16 + l15
    const int row0 = mt * 16 + hf * 8;
#pragma unroll
    for (int j = 0; j < 4; ++j) {
        const int col = nt * 64 + j * 16 + l15;
        const float bv = bias ? bias[col] : 0.0f;
#pragma unroll
        for (int g = 0; g < 8; ++g) {
            const float v = acc[j][g] + bv;
            const size_t idx = (size_t)(row0 + g) * Ncols + col;
            if (outF) outF[idx] = v;
            if (outB) outB[idx] = f2bf(v);
        }
    }
}

// ---------------- fp32 -> bf16 convert ----------------
__global__ __launch_bounds__(256)
void k_cvt_bf16(const float* __restrict__ in, unsigned short* __restrict__ out, size_t n)
{
    size_t i = (size_t)blockIdx.x * blockDim.x + threadIdx.x;
    if (i < n) out[i] = f2bf(in[i]);
}

// ---------------- im2col for 16x16 stride-16 patches (to bf16) ----------------
__global__ __launch_bounds__(256)
void k_im2col(const float* __restrict__ x, unsigned short* __restrict__ pat)
{
    size_t idx = (size_t)blockIdx.x * blockDim.x + threadIdx.x;
    if (idx >= (size_t)PTOK * KPATCH) return;
    int col = (int)(idx % KPATCH);
    int row = (int)(idx / KPATCH);
    int b = row / 196, p = row % 196;
    int ph = p / 14, pwi = p % 14;
    int c = col / 256, rem = col % 256;
    int kh = rem / 16, kw = rem % 16;
    float v = x[(((size_t)(b * 3 + c) * 224) + ph * 16 + kh) * 224 + pwi * 16 + kw];
    pat[idx] = f2bf(v);
}

// ---------------- token assembly: cls + patches + pos, zero pad rows ----------------
__global__ __launch_bounds__(256)
void k_assemble(const float* __restrict__ pout, const float* __restrict__ cls,
                const float* __restrict__ pos, float* __restrict__ t)
{
    size_t idx = (size_t)blockIdx.x * blockDim.x + threadIdx.x;
    if (idx >= (size_t)RPAD * D_) return;
    int d = (int)(idx % D_);
    int r = (int)(idx / D_);
    float v = 0.0f;
    if (r < RREAL) {
        int b = r / M_, m = r % M_;
        if (m == 0) v = cls[d] + pos[d];
        else        v = pout[(size_t)(b * 196 + m - 1) * D_ + d] + pos[(size_t)m * D_ + d];
    }
    t[idx] = v;
}

// ---------------- LayerNorm (wave per row, shuffle reduction) ----------------
__global__ __launch_bounds__(256)
void k_layernorm(const float* __restrict__ in, const float* __restrict__ g,
                 const float* __restrict__ bta, float* __restrict__ outF,
                 unsigned short* __restrict__ outB, int rows)
{
    int warp = (int)(threadIdx.x >> 5);
    int lane = (int)(threadIdx.x & 31);
    int row = (int)blockIdx.x * 8 + warp;
    if (row >= rows) return;
    const float* p = in + (size_t)row * D_;
    float s = 0.f, s2 = 0.f;
    for (int i = lane; i < D_; i += 32) { float v = p[i]; s += v; s2 += v * v; }
    for (int m = 16; m >= 1; m >>= 1) { s += __shfl_xor(s, m); s2 += __shfl_xor(s2, m); }
    float mean = s * (1.0f / D_);
    float var  = s2 * (1.0f / D_) - mean * mean;
    float inv  = rsqrtf(var + 1e-5f);
    for (int i = lane; i < D_; i += 32) {
        float v = (p[i] - mean) * inv * g[i] + bta[i];
        size_t idx = (size_t)row * D_ + i;
        if (outF) outF[idx] = v;
        if (outB) outB[idx] = f2bf(v);
    }
}

// ---------------- depthwise conv K=3 over tokens + SiLU ----------------
__global__ __launch_bounds__(256)
void k_dwconv_silu(const float* __restrict__ xp, const float* __restrict__ w,
                   const float* __restrict__ bias, float* __restrict__ oF,
                   unsigned short* __restrict__ oB)
{
    size_t idx = (size_t)blockIdx.x * blockDim.x + threadIdx.x;
    if (idx >= (size_t)RPAD * E_) return;
    int e = (int)(idx % E_);
    int r = (int)(idx / E_);
    float vO = 0.0f;
    if (r < RREAL) {
        int b = r / M_, m = r % M_;
        float s = bias[e];
#pragma unroll
        for (int k = 0; k < 3; ++k) {
            int mm = m + k - 1;
            if (mm >= 0 && mm < M_)
                s += xp[(size_t)(b * M_ + mm) * E_ + e] * w[e * 3 + k];
        }
        vO = s / (1.0f + expf(-s));     // SiLU
    }
    oF[idx] = vO;
    oB[idx] = f2bf(vO);
}

// ---------------- delta = softplus(proj + dt) (in-place) ----------------
__global__ __launch_bounds__(256)
void k_softplus_dt(float* __restrict__ a, const float* __restrict__ dt)
{
    size_t idx = (size_t)blockIdx.x * blockDim.x + threadIdx.x;
    if (idx >= (size_t)RPAD * E_) return;
    int e = (int)(idx % E_);
    float d = a[idx] + dt[e];
    a[idx] = (d > 20.0f) ? d : log1pf(expf(d));
}

// ---------------- selective scan: lane per (b,e,n), shuffle-reduce over n ----
__global__ __launch_bounds__(256)
void k_scan(const unsigned short* __restrict__ Bm, const unsigned short* __restrict__ Cm,
            const float* __restrict__ del, const float* __restrict__ Ap,
            const float* __restrict__ X, float* __restrict__ Y, int rev)
{
    int t = (int)(blockIdx.x * blockDim.x + threadIdx.x);   // 8*768*16 = 98304
    int n = t & 15;
    int idx2 = t >> 4;
    int e = idx2 % E_;
    int b = idx2 / E_;
    if (b >= B_) return;
    const float a_coef = Ap[e * N_ + n];
    float h = 0.0f;
    for (int s = 0; s < M_; ++s) {
        int m = rev ? (M_ - 1 - s) : s;
        size_t r = (size_t)(b * M_ + m);
        float dl = del[r * E_ + e];
        float xv = X[r * E_ + e];
        float bm = bf2f(Bm[r * EN_ + (size_t)e * N_ + n]) * dl;
        h = (dl * a_coef) * h + bm * xv;
        float c = bf2f(Cm[r * EN_ + (size_t)e * N_ + n]);
        float p = h * c;
        p += __shfl_xor(p, 1);
        p += __shfl_xor(p, 2);
        p += __shfl_xor(p, 4);
        p += __shfl_xor(p, 8);
        if (n == 0) Y[r * E_ + e] = p;
    }
}

// ---------------- gate: y = (yf + yb) * silu(z) -> bf16 ----------------
__global__ __launch_bounds__(256)
void k_combine(const float* __restrict__ yf, const float* __restrict__ yb,
               const float* __restrict__ z, unsigned short* __restrict__ yB)
{
    size_t idx = (size_t)blockIdx.x * blockDim.x + threadIdx.x;
    if (idx >= (size_t)RPAD * E_) return;
    int r = (int)(idx / E_);
    float v = 0.0f;
    if (r < RREAL) {
        float zz = z[idx];
        v = (yf[idx] + yb[idx]) * (zz / (1.0f + expf(-zz)));
    }
    yB[idx] = f2bf(v);
}

// ---------------- residual add ----------------
__global__ __launch_bounds__(256)
void k_add(float* __restrict__ t, const float* __restrict__ yo, size_t n)
{
    size_t idx = (size_t)blockIdx.x * blockDim.x + threadIdx.x;
    if (idx < n) t[idx] += yo[idx];
}

// ===========================================================================
extern "C" void kernel_launch(void* const* d_in, const int* in_sizes, int n_in,
                              void* d_out, int out_size, void* d_ws, size_t ws_size,
                              hipStream_t stream)
{
    (void)in_sizes; (void)n_in; (void)out_size; (void)ws_size;

    const float* x    = (const float*)d_in[0];
    const float* pw   = (const float*)d_in[1];
    const float* pb   = (const float*)d_in[2];
    const float* cls  = (const float*)d_in[3];
    const float* pos  = (const float*)d_in[4];
    const float* ln_g = (const float*)d_in[5];
    const float* ln_b = (const float*)d_in[6];
    const float* wx   = (const float*)d_in[7];
    const float* bx   = (const float*)d_in[8];
    const float* wz   = (const float*)d_in[9];
    const float* bz   = (const float*)d_in[10];
    const float* kwf  = (const float*)d_in[11];
    const float* kbf  = (const float*)d_in[12];
    const float* kwb  = (const float*)d_in[13];
    const float* kbb  = (const float*)d_in[14];
    const float* bfw  = (const float*)d_in[15];
    const float* bfb  = (const float*)d_in[16];
    const float* cfw  = (const float*)d_in[17];
    const float* cfb  = (const float*)d_in[18];
    const float* dfw  = (const float*)d_in[19];
    const float* dfb  = (const float*)d_in[20];
    const float* dtf  = (const float*)d_in[21];
    const float* af   = (const float*)d_in[22];
    const float* bbw  = (const float*)d_in[23];
    const float* bbb  = (const float*)d_in[24];
    const float* cbw  = (const float*)d_in[25];
    const float* cbb  = (const float*)d_in[26];
    const float* dbw  = (const float*)d_in[27];
    const float* dbb  = (const float*)d_in[28];
    const float* dtb  = (const float*)d_in[29];
    const float* ab   = (const float*)d_in[30];
    const float* wo   = (const float*)d_in[31];
    const float* bo   = (const float*)d_in[32];
    const float* fg   = (const float*)d_in[33];
    const float* fbv  = (const float*)d_in[34];

    // ---- workspace allocator ----
    char* base = (char*)d_ws;
    size_t off = 0;
    auto alloc = [&](size_t bytes) -> char* {
        off = (off + 255) & ~(size_t)255;
        char* p = base + off;
        off += bytes;
        return p;
    };

    // bf16 weights
    unsigned short* pwB  = (unsigned short*)alloc((size_t)D_ * KPATCH * 2);
    unsigned short* wxB  = (unsigned short*)alloc((size_t)L_ * E_ * D_ * 2);
    unsigned short* wzB  = (unsigned short*)alloc((size_t)L_ * E_ * D_ * 2);
    unsigned short* bfwB = (unsigned short*)alloc((size_t)L_ * EN_ * E_ * 2);
    unsigned short* cfwB = (unsigned short*)alloc((size_t)L_ * EN_ * E_ * 2);
    unsigned short* bbwB = (unsigned short*)alloc((size_t)L_ * EN_ * E_ * 2);
    unsigned short* cbwB = (unsigned short*)alloc((size_t)L_ * EN_ * E_ * 2);
    unsigned short* dfwB = (unsigned short*)alloc((size_t)L_ * E_ * E_ * 2);
    unsigned short* dbwB = (unsigned short*)alloc((size_t)L_ * E_ * E_ * 2);
    unsigned short* woB  = (unsigned short*)alloc((size_t)L_ * D_ * E_ * 2);

    // activations
    unsigned short* patB = (unsigned short*)alloc((size_t)PTOK * KPATCH * 2);
    float* pout = (float*)alloc((size_t)PTOK * D_ * 4);
    float* t    = (float*)alloc((size_t)RPAD * D_ * 4);
    unsigned short* xnB = (unsigned short*)alloc((size_t)RPAD * D_ * 2);
    float* xp   = (float*)alloc((size_t)RPAD * E_ * 4);
    float* zbuf = (float*)alloc((size_t)RPAD * E_ * 4);
    float* xfF  = (float*)alloc((size_t)RPAD * E_ * 4);
    unsigned short* xfB = (unsigned short*)alloc((size_t)RPAD * E_ * 2);
    float* xbF  = (float*)alloc((size_t)RPAD * E_ * 4);
    unsigned short* xbB = (unsigned short*)alloc((size_t)RPAD * E_ * 2);
    unsigned short* BfB = (unsigned short*)alloc((size_t)RPAD * EN_ * 2);
    unsigned short* CfB = (unsigned short*)alloc((size_t)RPAD * EN_ * 2);
    unsigned short* BbB = (unsigned short*)alloc((size_t)RPAD * EN_ * 2);
    unsigned short* CbB = (unsigned short*)alloc((size_t)RPAD * EN_ * 2);
    float* delF = (float*)alloc((size_t)RPAD * E_ * 4);
    float* delB = (float*)alloc((size_t)RPAD * E_ * 4);
    float* yfv  = (float*)alloc((size_t)RPAD * E_ * 4);
    float* ybv  = (float*)alloc((size_t)RPAD * E_ * 4);
    unsigned short* ycB = (unsigned short*)alloc((size_t)RPAD * E_ * 2);
    float* yo   = (float*)alloc((size_t)RPAD * D_ * 4);

    auto grid1 = [](size_t n) -> unsigned { return (unsigned)((n + 255) / 256); };
    auto cvt = [&](const float* s, unsigned short* d, size_t n) {
        k_cvt_bf16<<<grid1(n), 256, 0, stream>>>(s, d, n);
    };
    auto gemm = [&](const unsigned short* A, const unsigned short* Wm, const float* bi,
                    float* oF, unsigned short* oB, int Mt, int Nt, int K) {
        int waves = Mt * Nt;
        k_gemm_wmma<<<(waves + 7) / 8, 256, 0, stream>>>(A, Wm, bi, oF, oB, Mt, Nt, K);
    };

    // ---- weight conversion (every call; deterministic) ----
    cvt(pw,  pwB,  (size_t)D_ * KPATCH);
    cvt(wx,  wxB,  (size_t)L_ * E_ * D_);
    cvt(wz,  wzB,  (size_t)L_ * E_ * D_);
    cvt(bfw, bfwB, (size_t)L_ * EN_ * E_);
    cvt(cfw, cfwB, (size_t)L_ * EN_ * E_);
    cvt(bbw, bbwB, (size_t)L_ * EN_ * E_);
    cvt(cbw, cbwB, (size_t)L_ * EN_ * E_);
    cvt(dfw, dfwB, (size_t)L_ * E_ * E_);
    cvt(dbw, dbwB, (size_t)L_ * E_ * E_);
    cvt(wo,  woB,  (size_t)L_ * D_ * E_);

    // ---- patch embed (im2col + WMMA GEMM) ----
    k_im2col<<<grid1((size_t)PTOK * KPATCH), 256, 0, stream>>>(x, patB);
    gemm(patB, pwB, pb, pout, nullptr, PTOK / 16, D_ / 64, KPATCH);
    k_assemble<<<grid1((size_t)RPAD * D_), 256, 0, stream>>>(pout, cls, pos, t);

    // ---- layers ----
    for (int l = 0; l < L_; ++l) {
        k_layernorm<<<(RPAD + 7) / 8, 256, 0, stream>>>(
            t, ln_g + (size_t)l * D_, ln_b + (size_t)l * D_, nullptr, xnB, RPAD);

        gemm(xnB, wxB + (size_t)l * E_ * D_, bx + (size_t)l * E_, xp,   nullptr, RPAD / 16, E_ / 64, D_);
        gemm(xnB, wzB + (size_t)l * E_ * D_, bz + (size_t)l * E_, zbuf, nullptr, RPAD / 16, E_ / 64, D_);

        k_dwconv_silu<<<grid1((size_t)RPAD * E_), 256, 0, stream>>>(
            xp, kwf + (size_t)l * E_ * 3, kbf + (size_t)l * E_, xfF, xfB);
        k_dwconv_silu<<<grid1((size_t)RPAD * E_), 256, 0, stream>>>(
            xp, kwb + (size_t)l * E_ * 3, kbb + (size_t)l * E_, xbF, xbB);

        // forward branch projections
        gemm(xfB, bfwB + (size_t)l * EN_ * E_, bfb + (size_t)l * EN_, nullptr, BfB, RPAD / 16, EN_ / 64, E_);
        gemm(xfB, cfwB + (size_t)l * EN_ * E_, cfb + (size_t)l * EN_, nullptr, CfB, RPAD / 16, EN_ / 64, E_);
        gemm(xfB, dfwB + (size_t)l * E_ * E_,  dfb + (size_t)l * E_,  delF, nullptr, RPAD / 16, E_ / 64, E_);
        k_softplus_dt<<<grid1((size_t)RPAD * E_), 256, 0, stream>>>(delF, dtf + (size_t)l * E_);

        // backward branch projections
        gemm(xbB, bbwB + (size_t)l * EN_ * E_, bbb + (size_t)l * EN_, nullptr, BbB, RPAD / 16, EN_ / 64, E_);
        gemm(xbB, cbwB + (size_t)l * EN_ * E_, cbb + (size_t)l * EN_, nullptr, CbB, RPAD / 16, EN_ / 64, E_);
        gemm(xbB, dbwB + (size_t)l * E_ * E_,  dbb + (size_t)l * E_,  delB, nullptr, RPAD / 16, E_ / 64, E_);
        k_softplus_dt<<<grid1((size_t)RPAD * E_), 256, 0, stream>>>(delB, dtb + (size_t)l * E_);

        // selective scans (98304 lanes = 384 blocks)
        k_scan<<<384, 256, 0, stream>>>(BfB, CfB, delF, af + (size_t)l * E_ * N_, xfF, yfv, 0);
        k_scan<<<384, 256, 0, stream>>>(BbB, CbB, delB, ab + (size_t)l * E_ * N_, xbF, ybv, 1);

        k_combine<<<grid1((size_t)RPAD * E_), 256, 0, stream>>>(yfv, ybv, zbuf, ycB);
        gemm(ycB, woB + (size_t)l * D_ * E_, bo + (size_t)l * D_, yo, nullptr, RPAD / 16, D_ / 64, E_);
        k_add<<<grid1((size_t)RPAD * D_), 256, 0, stream>>>(t, yo, (size_t)RPAD * D_);
    }

    // ---- final LN -> d_out (real rows only: exactly out_size elements) ----
    k_layernorm<<<(RREAL + 7) / 8, 256, 0, stream>>>(
        t, fg, fbv, (float*)d_out, nullptr, RREAL);
}
